// BinaryDenseLayer_27522150433241
// MI455X (gfx1250) — matrix operand order repaired and checked by reading, not compiled, not run on hardware
//
#include <hip/hip_runtime.h>
#include <hip/hip_bf16.h>

typedef __attribute__((ext_vector_type(16))) _Float16 v16h;
typedef __attribute__((ext_vector_type(8)))  float    v8f;
typedef __attribute__((ext_vector_type(4)))  float    v4f;

#define K_DIM   512
#define N_DIM   128
#define KTILES  (K_DIM / 32)   // 16
#define NTILES  (N_DIM / 16)   // 8
#define FRAG_HALVES 512        // 32 lanes * 16 halves per (kt,nt) B fragment

// --------------------------------------------------------------------------
// Kernel 1: binarize W (f32, K x N row-major) into f16 B-fragments laid out
// exactly as V_WMMA_F32_16X16X32_F16 wants its B operand (ISA 7.12.2):
//   B tile is 32(K) x 16(N); lane L holds column N=L%16, 16 K halves in table order.
// Fragment storage: Bfrag[((kt*NTILES + nt)*32 + lane)*16 + h]
// --------------------------------------------------------------------------
__global__ __launch_bounds__(256) void binarizeW(const float* __restrict__ W,
                                                 _Float16* __restrict__ Bfrag) {
    int idx = blockIdx.x * blockDim.x + threadIdx.x;   // 0 .. 65535
    if (idx >= K_DIM * N_DIM) return;
    int k = idx >> 7;       // / N_DIM
    int n = idx & (N_DIM - 1);
    float w = W[idx];
    float s = (w > 0.0f) ? 1.0f : ((w < 0.0f) ? -1.0f : 0.0f);  // jnp.sign
    int kt   = k >> 5;      // K tile of 32
    int lane = (n & 15) + (((k >> 3) & 1) << 4);  // N col + 16*K half-group
    int nt   = n >> 4;      // N tile of 16
    int h    = (k & 7) + (((k >> 4) & 1) << 3);  // K in table order
    Bfrag[((size_t)(kt * NTILES + nt) * 32 + lane) * 16 + h] = (_Float16)s;
}

// Load all 8 B fragments of K-step (ktv) into register buffer `buf`.
#define LOAD_B(ktv, buf)                                                      \
    {                                                                         \
        const _Float16* bptr_ =                                               \
            Bfrag + (size_t)(ktv) * NTILES * FRAG_HALVES + lane * 16;         \
        _Pragma("unroll")                                                     \
        for (int nt_ = 0; nt_ < NTILES; ++nt_)                                \
            buf[nt_] = *(const v16h*)(bptr_ + nt_ * FRAG_HALVES);             \
    }

// Load + convert the A fragment of K-step (ktv) into v16h `areg`.
#define LOAD_A(ktv, areg)                                                     \
    {                                                                         \
        const int kb_ = (ktv) * 32;                                           \
        v4f a0_ = *(const v4f*)(xrow + kb_ + klo + 0);                        \
        v4f a1_ = *(const v4f*)(xrow + kb_ + klo + 4);                        \
        v4f a2_ = *(const v4f*)(xrow + kb_ + 16 + klo + 0);                   \
        v4f a3_ = *(const v4f*)(xrow + kb_ + 16 + klo + 4);                   \
        _Pragma("unroll")                                                     \
        for (int i_ = 0; i_ < 4; ++i_) {                                      \
            areg[i_ +  0] = (_Float16)a0_[i_];                                \
            areg[i_ +  4] = (_Float16)a1_[i_];                                \
            areg[i_ +  8] = (_Float16)a2_[i_];                                \
            areg[i_ + 12] = (_Float16)a3_[i_];                                \
        }                                                                     \
    }

// 8 back-to-back WMMAs sharing one A fragment.
#define WMMA_CHAIN(areg, buf)                                                 \
    {                                                                         \
        _Pragma("unroll")                                                     \
        for (int nt_ = 0; nt_ < NTILES; ++nt_)                                \
            acc[nt_] = __builtin_amdgcn_wmma_f32_16x16x32_f16(                \
                false, areg, false, buf[nt_], (short)0, acc[nt_],             \
                false, false);                                                \
    }

// --------------------------------------------------------------------------
// Kernel 2: out[M=65536, N=128] = x[M,512] @ Bfrag + bias
// One wave computes a 16(M) x 128(N) strip: 8 f32 accumulator tiles.
// A fragment straight from global x (f32 -> f16):
//   lane L (<16):  M = m0+L,   halves = K {kb+0..7, kb+16..23}
//   lane L (>=16): M = m0+L-16, halves = K {kb+8..15, kb+24..31}
// B fragments are explicitly double-buffered across K-steps (ping-pong
// register buffers, prefetch issued before the consuming WMMA chain) so the
// 8 WMMAs per step issue without per-tile load waits.
// --------------------------------------------------------------------------
__global__ __launch_bounds__(256) void bin_gemm_wmma(
        const float*    __restrict__ x,
        const _Float16* __restrict__ Bfrag,
        const float*    __restrict__ bias,
        float*          __restrict__ out) {
    const int lane = threadIdx.x & 31;
    const int wave = threadIdx.x >> 5;                 // 0..7
    const int m0   = (blockIdx.x * 8 + wave) * 16;     // wave's row base
    const int row  = m0 + (lane & 15);
    const int klo  = (lane >> 4) << 3;                 // 0 or 8

    v8f acc[NTILES];
#pragma unroll
    for (int nt = 0; nt < NTILES; ++nt) acc[nt] = (v8f)(0.0f);

    const float* xrow = x + (size_t)row * K_DIM;

    v16h b0[NTILES], b1[NTILES];
    LOAD_B(0, b0);                       // prologue: fill buffer 0

#pragma unroll 1
    for (int kt = 0; kt < KTILES; kt += 2) {
        // Prefetch next step's B while this step's WMMAs run.
        LOAD_B(kt + 1, b1);
        v16h a0;
        LOAD_A(kt, a0);
        WMMA_CHAIN(a0, b0);

        if (kt + 2 < KTILES) { LOAD_B(kt + 2, b0); }
        v16h a1;
        LOAD_A(kt + 1, a1);
        WMMA_CHAIN(a1, b1);
    }

    // Store: C/D layout -> VGPR v, lane L: M = m0 + v + (L>=16 ? 8 : 0),
    //                                   N = nt*16 + (L & 15)
    const int col   = lane & 15;
    const int rbase = m0 + ((lane >> 4) << 3);
#pragma unroll
    for (int nt = 0; nt < NTILES; ++nt) {
        const float bv = bias[nt * 16 + col];
        float* op = out + (size_t)rbase * N_DIM + nt * 16 + col;
#pragma unroll
        for (int v = 0; v < 8; ++v) {
            op[(size_t)v * N_DIM] = acc[nt][v] + bv;
        }
    }
}

extern "C" void kernel_launch(void* const* d_in, const int* in_sizes, int n_in,
                              void* d_out, int out_size, void* d_ws, size_t ws_size,
                              hipStream_t stream) {
    const float* x = (const float*)d_in[0];   // (65536, 512)
    const float* W = (const float*)d_in[1];   // (512, 128)
    const float* b = (const float*)d_in[2];   // (128,)
    float* out = (float*)d_out;               // (65536, 128)

    _Float16* Bfrag = (_Float16*)d_ws;        // 512*128*2 = 128 KB scratch

    binarizeW<<<(K_DIM * N_DIM + 255) / 256, 256, 0, stream>>>(W, Bfrag);
    // 65536 rows / (8 waves * 16 rows) = 512 blocks
    bin_gemm_wmma<<<512, 256, 0, stream>>>(x, Bfrag, b, out);
}
